// GRU4Rec_34076270526649
// MI455X (gfx1250) — matrix operand order repaired and checked by reading, not compiled, not run on hardware
//
#include <hip/hip_runtime.h>
#include <hip/hip_bf16.h>

typedef __attribute__((ext_vector_type(16))) _Float16 v16h;
typedef __attribute__((ext_vector_type(8)))  float    v8f;

#define BS 128
#define SEQ 100
#define DD 64
#define TH1 0.007f
#define TH2 0.007f

// ---------------------------------------------------------------------------
// WMMA helpers: CDNA5 v_wmma_f32_16x16x32_f16 operand loaders (row-major src)
// A layout (16x32, MxK): lane m=lane&15 holds row m; half=lane>>4 adds K+8.
//   VGPR v holds K pair: (v&3)*2 + (v>>2)*16 (+half*8).
// B layout (32x16, KxN): lane n=lane&15 holds col n (= row n of B^T); half
//   adds K+8; VGPR v holds K pair (v&3)*2 + (v>>2)*16 (same as A).
// ---------------------------------------------------------------------------
__device__ __forceinline__ v16h load_matA(const _Float16* base, int ld) {
  const int lane = threadIdx.x & 31;
  const int m = lane & 15, hf = lane >> 4;
  const _Float16* rp = base + m * ld + hf * 8;
  union { v16h v; unsigned u[8]; } r;
#pragma unroll
  for (int v = 0; v < 8; ++v)
    r.u[v] = *(const unsigned*)(rp + ((v & 3) * 2 + (v >> 2) * 16));
  return r.v;
}

__device__ __forceinline__ v16h load_matB(const _Float16* base, int ld) {
  const int lane = threadIdx.x & 31;
  const int n = lane & 15, hf = lane >> 4;
  const _Float16* rp = base + n * ld + hf * 8;
  union { v16h v; unsigned u[8]; } r;
#pragma unroll
  for (int v = 0; v < 8; ++v)
    r.u[v] = *(const unsigned*)(rp + ((v & 3) * 2 + (v >> 2) * 16));
  return r.v;
}

__device__ __forceinline__ v8f wmma16(v16h a, v16h b, v8f c) {
  return __builtin_amdgcn_wmma_f32_16x16x32_f16(false, a, false, b,
                                                (short)0, c, false, false);
}

// ---------------------------------------------------------------------------
// Embedding gather: seq_d1/seq_d2 -> xcat f32 (first 100 rows) + f16 tiles,
// item+negatives -> f16 feature matrix for the predictor.
// ---------------------------------------------------------------------------
__global__ void k_gather(const int* i_node, const int* neg,
                         const int* seq1, const int* seq2, const float* emb,
                         float* xcat1, float* xcat2,
                         _Float16* s1h, _Float16* s2h, _Float16* ifh) {
  int idx = blockIdx.x * blockDim.x + threadIdx.x;
  const int total = BS * 300 * DD;
  if (idx >= total) return;
  int k = idx & 63;
  int row = idx >> 6;
  int b = row / 300, r = row % 300;
  if (r < 100) {
    int it = seq1[b * 100 + r];
    float v = emb[(size_t)it * DD + k];
    xcat1[(b * 200 + r) * DD + k] = v;
    s1h[(b * 112 + r) * DD + k] = (_Float16)v;
  } else if (r < 200) {
    int l = r - 100;
    int it = seq2[b * 100 + l];
    float v = emb[(size_t)it * DD + k];
    xcat2[(b * 200 + l) * DD + k] = v;
    s2h[(b * 112 + l) * DD + k] = (_Float16)v;
  } else {
    int l = r - 200;
    int it = (l == 0) ? i_node[b] : neg[b * 99 + (l - 1)];
    ifh[(b * 100 + l) * DD + k] = (_Float16)emb[(size_t)it * DD + k];
  }
}

__global__ void k_padzero(_Float16* p, int Lp, int L) {
  int idx = blockIdx.x * blockDim.x + threadIdx.x;
  int per = (Lp - L) * DD;
  if (idx >= BS * per) return;
  int b = idx / per, rem = idx % per;
  int l = L + rem / DD, k = rem & 63;
  p[(b * Lp + l) * DD + k] = (_Float16)0.f;
}

__global__ void k_f32tof16(const float* s, _Float16* d, int n) {
  int i = blockIdx.x * blockDim.x + threadIdx.x;
  if (i < n) d[i] = (_Float16)s[i];
}

// ---------------------------------------------------------------------------
// Per-batch gram maximum: g[b] = max_{l,m} A[b,l,:]·B[b,m,:]  (WMMA tiles)
// Interior tiles take an unmasked fast path (wave-uniform branch).
// ---------------------------------------------------------------------------
__global__ void k_gram_max(const _Float16* A, const _Float16* B,
                           int Lp, int L, int nt, float* g) {
  __shared__ float red[256];
  int b = blockIdx.x;
  const _Float16* Ab = A + (size_t)b * Lp * DD;
  const _Float16* Bb = B + (size_t)b * Lp * DD;
  int tid = threadIdx.x, wave = tid >> 5, lane = tid & 31;
  int m16 = lane & 15, hf = lane >> 4;
  float mx = -3.0e38f;
  for (int tidx = wave; tidx < nt * nt; tidx += 8) {
    int ti = tidx / nt, tj = tidx % nt;
    v8f acc = {};
#pragma unroll
    for (int ks = 0; ks < 2; ++ks) {
      v16h a = load_matA(Ab + ti * 16 * DD + ks * 32, DD);
      v16h bm = load_matB(Bb + tj * 16 * DD + ks * 32, DD);
      acc = wmma16(a, bm, acc);
    }
    if ((ti + 1) * 16 <= L && (tj + 1) * 16 <= L) {
      // interior tile: every element valid
#pragma unroll
      for (int r = 0; r < 8; ++r) mx = fmaxf(mx, acc[r]);
    } else {
      // edge tile: mask rows/cols beyond L (pad rows are zero-filled)
#pragma unroll
      for (int r = 0; r < 8; ++r) {
        int M = ti * 16 + r + hf * 8, N = tj * 16 + m16;
        if (M < L && N < L) mx = fmaxf(mx, acc[r]);
      }
    }
  }
  red[tid] = mx;
  __syncthreads();
  for (int s = 128; s > 0; s >>= 1) {
    if (tid < s) red[tid] = fmaxf(red[tid], red[tid + s]);
    __syncthreads();
  }
  if (tid == 0) g[b] = red[0];
}

// softmax over 128 batch values, threshold -> {0,1} mask
__global__ void k_softmax_mask(const float* g, float th, float* m) {
  __shared__ float red[128];
  int i = threadIdx.x;
  float v = g[i];
  red[i] = v; __syncthreads();
  for (int s = 64; s > 0; s >>= 1) {
    if (i < s) red[i] = fmaxf(red[i], red[i + s]);
    __syncthreads();
  }
  float mx = red[0]; __syncthreads();
  float e = __expf(v - mx);
  red[i] = e; __syncthreads();
  for (int s = 64; s > 0; s >>= 1) {
    if (i < s) red[i] += red[i + s];
    __syncthreads();
  }
  m[i] = (e / red[0] > th) ? 1.f : 0.f;
}

// t[l,k] = sum_b wbs[b]*m[b]*S[b,l,k]
__global__ void k_batchred(const float* S, int ldrow, int L,
                           const float* wbs, const float* msk, float* t) {
  int idx = blockIdx.x * blockDim.x + threadIdx.x;
  if (idx >= L * DD) return;
  int l = idx >> 6, k = idx & 63;
  float s = 0.f;
  for (int b = 0; b < BS; ++b)
    s += wbs[b] * msk[b] * S[(size_t)(b * ldrow + l) * DD + k];
  t[idx] = s;
}

// z[l,d] = t[l,:]·Wnn[d,:] + (sum wbs)*bnn[d] + bbs
__global__ void k_zgemm(const float* t, const float* Wnn, const float* bnn,
                        const float* bbs, const float* wbs, float* z, int L) {
  int idx = blockIdx.x * blockDim.x + threadIdx.x;
  if (idx >= L * DD) return;
  int l = idx >> 6, d = idx & 63;
  float a = 0.f;
  for (int k = 0; k < DD; ++k) a += t[l * DD + k] * Wnn[d * DD + k];
  float swb = 0.f;
  for (int b = 0; b < BS; ++b) swb += wbs[b];
  z[idx] = a + swb * bnn[d] + bbs[0];
}

// write broadcast z into second half of xcat, produce f16 mirror of xcat
__global__ void k_concat(float* xcat, _Float16* xcath, const float* z) {
  int idx = blockIdx.x * blockDim.x + threadIdx.x;
  if (idx >= BS * 200 * DD) return;
  int k = idx & 63;
  int l = (idx >> 6) % 200;
  float v;
  if (l >= 100) { v = z[(l - 100) * DD + k]; xcat[idx] = v; }
  else v = xcat[idx];
  xcath[idx] = (_Float16)v;
}

// xp[M,192] = Xh[M,64] @ Wih^T + bih   (M = 128*200, WMMA)
__global__ void k_xpgemm(const _Float16* X, const _Float16* Wh,
                         const float* bias, float* xp) {
  int wave = threadIdx.x >> 5, lane = threadIdx.x & 31;
  int gw = blockIdx.x * 8 + wave;
  int tm = gw / 12, tn = gw % 12;
  v8f acc = {};
#pragma unroll
  for (int ks = 0; ks < 2; ++ks) {
    v16h a = load_matA(X + tm * 16 * DD + ks * 32, DD);
    v16h bm = load_matB(Wh + tn * 16 * DD + ks * 32, DD);
    acc = wmma16(a, bm, acc);
  }
  int m16 = lane & 15, hf = lane >> 4;
  int col = tn * 16 + m16;
  float bv = bias[col];
#pragma unroll
  for (int r = 0; r < 8; ++r) {
    int row = tm * 16 + r + hf * 8;
    xp[(size_t)row * 192 + col] = acc[r] + bv;
  }
}

// ---------------------------------------------------------------------------
// Persistent GRU: one workgroup per domain (grid=2). h kept in LDS (f32+f16).
// 16 waves x 2 tasks; each task owns an aligned (r,z,n) gate tile triple so
// gate math happens in the WMMA output layout. Whh fragments, gate biases and
// all addressing are hoisted out of the 200-step serial loop: the per-step
// body is 2 LDS A-loads + 6 chained WMMAs + gate transcendentals per task.
// ---------------------------------------------------------------------------
__global__ __launch_bounds__(512) void k_gru(
    const float* xp1, const float* xp2,
    const _Float16* whh1, const _Float16* whh2,
    const float* bhh1, const float* bhh2,
    float* hseq1, float* hseq2, _Float16* hh1, _Float16* hh2) {
  const int dom = blockIdx.x;
  const float* xp = dom ? xp2 : xp1;
  const _Float16* whh = dom ? whh2 : whh1;
  const float* bhh = dom ? bhh2 : bhh1;
  float* hseq = dom ? hseq2 : hseq1;
  _Float16* hsh = dom ? hh2 : hh1;

  __shared__ float    sh[BS * DD];
  __shared__ _Float16 shh[BS * DD];
  int tid = threadIdx.x;
  for (int i = tid; i < BS * DD; i += 512) { sh[i] = 0.f; shh[i] = (_Float16)0.f; }

  int wave = tid >> 5, lane = tid & 31, m16 = lane & 15, hf = lane >> 4;

  // ---- loop-invariant preloads -------------------------------------------
  int ti_[2], abase_[2], d_[2];
  float br_[2], bz_[2], bn_[2];
  v16h bw[2][3][2];              // [task][gate][kstep] Whh fragments
  int hoff_[2][8];               // LDS offsets for h state
  int xbase_[2][8];              // xp row bases (row*192)
  int sbase_[2][8];              // hseq bases (row*200*64 + d)
  int pbase_[2][8];              // hsh bases  (row*208*64 + d)
#pragma unroll
  for (int q = 0; q < 2; ++q) {
    int task = wave * 2 + q;
    int ti = task >> 2, tj = task & 3;
    ti_[q] = ti;
    abase_[q] = ti * 16 * DD;
    int d = tj * 16 + m16;
    d_[q] = d;
    br_[q] = bhh[d]; bz_[q] = bhh[64 + d]; bn_[q] = bhh[128 + d];
#pragma unroll
    for (int gt = 0; gt < 3; ++gt) {
      int tn = gt * 4 + tj;
#pragma unroll
      for (int ks = 0; ks < 2; ++ks)
        bw[q][gt][ks] = load_matB(whh + tn * 16 * DD + ks * 32, DD);
    }
#pragma unroll
    for (int r = 0; r < 8; ++r) {
      int brow = ti * 16 + r + hf * 8;
      hoff_[q][r] = brow * DD + d;
      xbase_[q][r] = brow * 200 * 192;
      sbase_[q][r] = brow * 200 * DD + d;
      pbase_[q][r] = brow * 208 * DD + d;
    }
  }
  __syncthreads();

  for (int t = 0; t < 200; ++t) {
    float hnew[2][8];
#pragma unroll
    for (int q = 0; q < 2; ++q) {
      v16h a0 = load_matA(shh + abase_[q], DD);
      v16h a1 = load_matA(shh + abase_[q] + 32, DD);
      v8f c[3];
#pragma unroll
      for (int gt = 0; gt < 3; ++gt) {
        v8f acc = {};
        acc = wmma16(a0, bw[q][gt][0], acc);
        acc = wmma16(a1, bw[q][gt][1], acc);
        c[gt] = acc;
      }
      int d = d_[q];
#pragma unroll
      for (int r = 0; r < 8; ++r) {
        const float* xr = xp + xbase_[q][r] + t * 192;
        float rr = 1.f / (1.f + __expf(-(xr[d] + c[0][r] + br_[q])));
        float zz = 1.f / (1.f + __expf(-(xr[64 + d] + c[1][r] + bz_[q])));
        float nn = tanhf(xr[128 + d] + rr * (c[2][r] + bn_[q]));
        float hold = sh[hoff_[q][r]];
        hnew[q][r] = (1.f - zz) * nn + zz * hold;
      }
    }
    __syncthreads();
#pragma unroll
    for (int q = 0; q < 2; ++q) {
#pragma unroll
      for (int r = 0; r < 8; ++r) {
        float v = hnew[q][r];
        sh[hoff_[q][r]] = v;
        shh[hoff_[q][r]] = (_Float16)v;
        hseq[sbase_[q][r] + t * DD] = v;
        hsh[pbase_[q][r] + t * DD] = (_Float16)v;
      }
    }
    __syncthreads();
  }
}

// u[b,d] = (sum_l hseq[b,l,d] + sum_l z[l,d]) / 400
__global__ void k_mean(const float* hseq, const float* z,
                       float* u, _Float16* uh) {
  int idx = blockIdx.x * blockDim.x + threadIdx.x;
  if (idx >= BS * DD) return;
  int b = idx >> 6, d = idx & 63;
  float s = 0.f;
  for (int l = 0; l < 200; ++l) s += hseq[((size_t)b * 200 + l) * DD + d];
  for (int l = 0; l < 200; ++l) s += z[l * DD + d];
  float v = s * (1.f / 400.f);
  u[idx] = v;
  uh[idx] = (_Float16)v;
}

// ---------------------------------------------------------------------------
// Predictor: h = relu([u|i] @ fc1^T + b1) via WMMA (K=128), then fc2 dot +
// sigmoid via in-register half-wave butterfly reduction.
// ---------------------------------------------------------------------------
__global__ void k_pred(const _Float16* u1h, const _Float16* u2h,
                       const _Float16* ifh, const _Float16* fc1h,
                       const float* fc1b, const float* fc2w,
                       const float* fc2b, float* out) {
  int blk = blockIdx.x;
  int dom = blk / 100;
  const _Float16* uh = dom ? u2h : u1h;
  float* op = out + dom * 12800;
  int wave = threadIdx.x >> 5, lane = threadIdx.x & 31;
  int m16 = lane & 15, hf = lane >> 4;
  int rowtile = (blk % 100) * 8 + wave;
  int rowbase = rowtile * 16;

  v16h a[4];
  {
    int r = rowbase + m16;
    int b = r / 100;
    const _Float16* up = uh + b * DD;
    const _Float16* ip = ifh + (size_t)r * DD;
#pragma unroll
    for (int ks = 0; ks < 4; ++ks) {
      const _Float16* rp = (ks < 2 ? up + ks * 32 : ip + (ks - 2) * 32) + hf * 8;
      union { v16h v; unsigned w[8]; } tv;
#pragma unroll
      for (int v = 0; v < 8; ++v)
        tv.w[v] = *(const unsigned*)(rp + ((v & 3) * 2 + (v >> 2) * 16));
      a[ks] = tv.v;
    }
  }

  float part[8];
#pragma unroll
  for (int r = 0; r < 8; ++r) part[r] = 0.f;
#pragma unroll
  for (int ct = 0; ct < 8; ++ct) {
    v8f acc = {};
#pragma unroll
    for (int ks = 0; ks < 4; ++ks)
      acc = wmma16(a[ks], load_matB(fc1h + ct * 16 * 128 + ks * 32, 128), acc);
    float bv = fc1b[ct * 16 + m16];
    float w2 = fc2w[ct * 16 + m16];
#pragma unroll
    for (int r = 0; r < 8; ++r) {
      float h = acc[r] + bv;
      h = h > 0.f ? h : 0.f;
      part[r] += h * w2;
    }
  }
#pragma unroll
  for (int off = 1; off < 16; off <<= 1)
#pragma unroll
    for (int r = 0; r < 8; ++r)
      part[r] += __shfl_xor(part[r], off, 32);
  if (m16 == 0) {
#pragma unroll
    for (int r = 0; r < 8; ++r) {
      int row = rowbase + r + hf * 8;
      op[row] = 1.f / (1.f + __expf(-(part[r] + fc2b[0])));
    }
  }
}

// ---------------------------------------------------------------------------
static inline size_t al256(size_t x) { return (x + 255) & ~(size_t)255; }

extern "C" void kernel_launch(void* const* d_in, const int* in_sizes, int n_in,
                              void* d_out, int out_size, void* d_ws, size_t ws_size,
                              hipStream_t stream) {
  (void)in_sizes; (void)n_in; (void)out_size; (void)ws_size;
  const int* i_node = (const int*)d_in[1];
  const int* neg = (const int*)d_in[2];
  const int* seq1 = (const int*)d_in[3];
  const int* seq2 = (const int*)d_in[4];
  const float* emb = (const float*)d_in[7];
  const float *inc1_Wnn = (const float*)d_in[8],  *inc1_bnn = (const float*)d_in[9],
              *inc1_Wbs = (const float*)d_in[10], *inc1_bbs = (const float*)d_in[11];
  const float *inc2_Wnn = (const float*)d_in[12], *inc2_bnn = (const float*)d_in[13],
              *inc2_Wbs = (const float*)d_in[14], *inc2_bbs = (const float*)d_in[15];
  const float *itc1_Wnn = (const float*)d_in[16], *itc1_bnn = (const float*)d_in[17],
              *itc1_Wbs = (const float*)d_in[18], *itc1_bbs = (const float*)d_in[19];
  const float *itc2_Wnn = (const float*)d_in[20], *itc2_bnn = (const float*)d_in[21],
              *itc2_Wbs = (const float*)d_in[22], *itc2_bbs = (const float*)d_in[23];
  const float *g1_Wih = (const float*)d_in[24], *g1_Whh = (const float*)d_in[25],
              *g1_bih = (const float*)d_in[26], *g1_bhh = (const float*)d_in[27];
  const float *g2_Wih = (const float*)d_in[28], *g2_Whh = (const float*)d_in[29],
              *g2_bih = (const float*)d_in[30], *g2_bhh = (const float*)d_in[31];
  const float *fc1_W = (const float*)d_in[32], *fc1_b = (const float*)d_in[33];
  const float *fc2_W = (const float*)d_in[34], *fc2_b = (const float*)d_in[35];
  float* out = (float*)d_out;

  // workspace carve-up
  char* p = (char*)d_ws;
  size_t off = 0;
  auto take = [&](size_t bytes) { char* r = p + off; off = al256(off + bytes); return r; };
  _Float16* s1h  = (_Float16*)take((size_t)BS * 112 * DD * 2);
  _Float16* s2h  = (_Float16*)take((size_t)BS * 112 * DD * 2);
  _Float16* ifh  = (_Float16*)take((size_t)BS * 100 * DD * 2);
  float*    xc1  = (float*)take((size_t)BS * 200 * DD * 4);
  float*    xc2  = (float*)take((size_t)BS * 200 * DD * 4);
  _Float16* xc1h = (_Float16*)take((size_t)BS * 200 * DD * 2);
  _Float16* xc2h = (_Float16*)take((size_t)BS * 200 * DD * 2);
  float*    xp1  = (float*)take((size_t)BS * 200 * 192 * 4);
  float*    xp2  = (float*)take((size_t)BS * 200 * 192 * 4);
  float*    hs1  = (float*)take((size_t)BS * 200 * DD * 4);
  float*    hs2  = (float*)take((size_t)BS * 200 * DD * 4);
  _Float16* h1h  = (_Float16*)take((size_t)BS * 208 * DD * 2);
  _Float16* h2h  = (_Float16*)take((size_t)BS * 208 * DD * 2);
  _Float16* wih1h = (_Float16*)take(192 * 64 * 2);
  _Float16* whh1h = (_Float16*)take(192 * 64 * 2);
  _Float16* wih2h = (_Float16*)take(192 * 64 * 2);
  _Float16* whh2h = (_Float16*)take(192 * 64 * 2);
  _Float16* fc1h  = (_Float16*)take(128 * 128 * 2);
  float* gv1 = (float*)take(BS * 4);
  float* gv2 = (float*)take(BS * 4);
  float* gvt = (float*)take(BS * 4);
  float* mm1 = (float*)take(BS * 4);
  float* mm2 = (float*)take(BS * 4);
  float* mmt = (float*)take(BS * 4);
  float* tbuf = (float*)take(200 * DD * 4);
  float* z_i1 = (float*)take(100 * DD * 4);
  float* z_i2 = (float*)take(100 * DD * 4);
  float* z_t1 = (float*)take(200 * DD * 4);
  float* z_t2 = (float*)take(200 * DD * 4);
  float* u1 = (float*)take(BS * DD * 4);
  float* u2 = (float*)take(BS * DD * 4);
  _Float16* u1h = (_Float16*)take(BS * DD * 2);
  _Float16* u2h = (_Float16*)take(BS * DD * 2);

  auto cdiv = [](int a, int b) { return (a + b - 1) / b; };

  // 1) weight conversions to f16
  k_f32tof16<<<cdiv(192 * 64, 256), 256, 0, stream>>>(g1_Wih, wih1h, 192 * 64);
  k_f32tof16<<<cdiv(192 * 64, 256), 256, 0, stream>>>(g1_Whh, whh1h, 192 * 64);
  k_f32tof16<<<cdiv(192 * 64, 256), 256, 0, stream>>>(g2_Wih, wih2h, 192 * 64);
  k_f32tof16<<<cdiv(192 * 64, 256), 256, 0, stream>>>(g2_Whh, whh2h, 192 * 64);
  k_f32tof16<<<cdiv(128 * 128, 256), 256, 0, stream>>>(fc1_W, fc1h, 128 * 128);

  // 2) gathers + tile padding
  k_gather<<<cdiv(BS * 300 * DD, 256), 256, 0, stream>>>(
      i_node, neg, seq1, seq2, emb, xc1, xc2, s1h, s2h, ifh);
  k_padzero<<<cdiv(BS * 12 * DD, 256), 256, 0, stream>>>(s1h, 112, 100);
  k_padzero<<<cdiv(BS * 12 * DD, 256), 256, 0, stream>>>(s2h, 112, 100);
  k_padzero<<<cdiv(BS * 8 * DD, 256), 256, 0, stream>>>(h1h, 208, 200);
  k_padzero<<<cdiv(BS * 8 * DD, 256), 256, 0, stream>>>(h2h, 208, 200);

  // 3) InnerComp: gram max -> softmax mask -> z, then concat + f16 mirror
  k_gram_max<<<BS, 256, 0, stream>>>(s1h, s1h, 112, 100, 7, gv1);
  k_gram_max<<<BS, 256, 0, stream>>>(s2h, s2h, 112, 100, 7, gv2);
  k_softmax_mask<<<1, 128, 0, stream>>>(gv1, TH1, mm1);
  k_softmax_mask<<<1, 128, 0, stream>>>(gv2, TH1, mm2);
  k_batchred<<<cdiv(100 * DD, 256), 256, 0, stream>>>(xc1, 200, 100, inc1_Wbs, mm1, tbuf);
  k_zgemm<<<cdiv(100 * DD, 256), 256, 0, stream>>>(tbuf, inc1_Wnn, inc1_bnn, inc1_bbs, inc1_Wbs, z_i1, 100);
  k_batchred<<<cdiv(100 * DD, 256), 256, 0, stream>>>(xc2, 200, 100, inc2_Wbs, mm2, tbuf);
  k_zgemm<<<cdiv(100 * DD, 256), 256, 0, stream>>>(tbuf, inc2_Wnn, inc2_bnn, inc2_bbs, inc2_Wbs, z_i2, 100);
  k_concat<<<cdiv(BS * 200 * DD, 256), 256, 0, stream>>>(xc1, xc1h, z_i1);
  k_concat<<<cdiv(BS * 200 * DD, 256), 256, 0, stream>>>(xc2, xc2h, z_i2);

  // 4) GRU input projections (big WMMA GEMM) + persistent recurrent kernel
  k_xpgemm<<<2400, 256, 0, stream>>>(xc1h, wih1h, g1_bih, xp1);
  k_xpgemm<<<2400, 256, 0, stream>>>(xc2h, wih2h, g2_bih, xp2);
  k_gru<<<2, 512, 0, stream>>>(xp1, xp2, whh1h, whh2h, g1_bhh, g2_bhh,
                               hs1, hs2, h1h, h2h);

  // 5) InterComp (cross gram is symmetric in max -> one mask for both)
  k_gram_max<<<BS, 256, 0, stream>>>(h1h, h2h, 208, 200, 13, gvt);
  k_softmax_mask<<<1, 128, 0, stream>>>(gvt, TH2, mmt);
  k_batchred<<<cdiv(200 * DD, 256), 256, 0, stream>>>(hs2, 200, 200, itc1_Wbs, mmt, tbuf);
  k_zgemm<<<cdiv(200 * DD, 256), 256, 0, stream>>>(tbuf, itc1_Wnn, itc1_bnn, itc1_bbs, itc1_Wbs, z_t1, 200);
  k_batchred<<<cdiv(200 * DD, 256), 256, 0, stream>>>(hs1, 200, 200, itc2_Wbs, mmt, tbuf);
  k_zgemm<<<cdiv(200 * DD, 256), 256, 0, stream>>>(tbuf, itc2_Wnn, itc2_bnn, itc2_bbs, itc2_Wbs, z_t2, 200);

  // 6) user vectors + predictor
  k_mean<<<cdiv(BS * DD, 256), 256, 0, stream>>>(hs1, z_t1, u1, u1h);
  k_mean<<<cdiv(BS * DD, 256), 256, 0, stream>>>(hs2, z_t2, u2, u2h);
  k_pred<<<200, 256, 0, stream>>>(u1h, u2h, ifh, fc1h, fc1_b, fc2_W, fc2_b, out);
}
